// ACTP_68143951119114
// MI455X (gfx1250) — compile-verified, matrix operand order checked by reading, not run on hardware
//
#include <hip/hip_runtime.h>

#define H   200
#define DT  48
#define DA  6
#define TSTEPS 20
#define NTHREADS 512   // 16 waves
#define BT 32          // batch rows per block (2 WMMA M-tiles)

typedef __attribute__((ext_vector_type(16))) _Float16 v16h;
typedef __attribute__((ext_vector_type(8)))  float    v8f;

// Inverse map: (row m in tile, K-index k) -> flat index in ISA 16-bit A-matrix
// fragment order. Per 32-wide K tile: lanes 0-15 hold M=0..15 with K {0..7,16..23},
// lanes 16-31 hold K {8..15,24..31}; halves j=2v+p per VGPR pair.
__device__ __forceinline__ int afrag_idx(int m, int k) {
  int kt = k >> 5;
  int kl = k & 31;
  int lh = (kl >> 3) & 1;
  int kb = kl - (lh << 3);
  int v, p;
  if (kb < 8) { v = kb >> 1; p = kb & 1; }
  else        { v = 4 + ((kb - 16) >> 1); p = kb & 1; }
  int j = 2 * v + p;
  int lane = (lh << 4) | m;
  return kt * 512 + lane * 16 + j;
}

__device__ __forceinline__ float sigf(float x) { return 1.0f / (1.0f + __expf(-x)); }

// Dual-M-tile K loop: one B fragment (global, L2-hot) feeds TWO wmma ops
// (A fragments for M-tile 0 and 1 from LDS). Doubles FLOPs per byte of L2 traffic.
__device__ __forceinline__ void wmma_k2(const _Float16* af, int mstride,
                                        const _Float16* __restrict__ wf, int nkt,
                                        v8f& acc0, v8f& acc1, int lane) {
  for (int kt = 0; kt < nkt; ++kt) {
    if (kt + 1 < nkt) __builtin_prefetch(wf + (kt + 1) * 512 + lane * 16, 0, 1);
    v16h b  = *(const v16h*)(wf + kt * 512 + lane * 16);
    v16h a0 = *(const v16h*)(af + kt * 512 + lane * 16);
    v16h a1 = *(const v16h*)(af + mstride + kt * 512 + lane * 16);
    acc0 = __builtin_amdgcn_wmma_f32_16x16x32_f16(false, a0, false, b, (short)0, acc0,
                                                  false, false);
    acc1 = __builtin_amdgcn_wmma_f32_16x16x32_f16(false, a1, false, b, (short)0, acc1,
                                                  false, false);
  }
}

// Pre-swizzle W (row-major [N_out, K_in]) into fragments [nt][kt][lane][16 halves]
// matching the 16-bit B-matrix (K x N) per-lane layout; zero-pad out-of-range.
__global__ void prep_wfrag(const float* __restrict__ W, _Float16* __restrict__ dst,
                           int N, int K, int KT, int total) {
  int id = blockIdx.x * 256 + threadIdx.x;
  if (id >= total) return;
  int j    = id & 15;
  int lane = (id >> 4) & 31;
  int kt   = (id >> 9) % KT;
  int nt   = id / (512 * KT);
  int n  = nt * 16 + (lane & 15);
  int kh = (lane >> 4) * 8;
  int v = j >> 1, p = j & 1;
  int kl = (v < 4) ? (kh + 2 * v + p) : (16 + kh + 2 * (v - 4) + p);
  int k = kt * 32 + kl;
  float val = (n < N && k < K) ? W[(size_t)n * K + k] : 0.0f;
  dst[id] = (_Float16)val;
}

__global__ void prep_bias(const float* __restrict__ bih1, const float* __restrict__ bhh1,
                          const float* __restrict__ bih2, const float* __restrict__ bhh2,
                          const float* __restrict__ b1,   const float* __restrict__ b2,
                          float* __restrict__ bb1, float* __restrict__ bb2,
                          float* __restrict__ b1p, float* __restrict__ b2p) {
  int id = blockIdx.x * 256 + threadIdx.x;
  if (id < 800) { bb1[id] = bih1[id] + bhh1[id]; bb2[id] = bih2[id] + bhh2[id]; }
  if (id < 208) b1p[id] = (id < 200) ? b1[id] : 0.0f;
  if (id < 48)  b2p[id] = b2[id];
}

__global__ __launch_bounds__(NTHREADS, 1) void actp_main(
    const float* __restrict__ tactiles, const float* __restrict__ actions,
    const int* __restrict__ cfp,
    const _Float16* __restrict__ Wih1f, const _Float16* __restrict__ Whh1f,
    const _Float16* __restrict__ Wih2f, const _Float16* __restrict__ Whh2f,
    const _Float16* __restrict__ W1f,  const _Float16* __restrict__ W2f,
    const float* __restrict__ bb1, const float* __restrict__ bb2,
    const float* __restrict__ b1p, const float* __restrict__ b2p,
    float* __restrict__ out, int Bk)
{
  __shared__ float zbuf[BT * 800];          // gate pre-activations, f32  (100 KB)
  __shared__ float c1s[BT * H];
  __shared__ float c2s[BT * H];
  __shared__ float x1raw[BT * DT];
  __shared__ float o4raw[BT * DT];
  __shared__ float actraw[BT * DA];
  __shared__ float st0raw[BT * DA];
  // activation fragments, [mt][kt][lane][16 halves]; mt stride = nkt*512
  __shared__ __align__(32) _Float16 x1f[2 * 2 * 512];   // x1 padded K=64
  __shared__ __align__(32) _Float16 h1f[2 * 7 * 512];   // h1 padded K=224
  __shared__ __align__(32) _Float16 x2f[2 * 8 * 512];   // [h1, tiled] padded K=256
  __shared__ __align__(32) _Float16 h2f[2 * 7 * 512];   // h2 padded K=224
  __shared__ __align__(32) _Float16 lpf[2 * 8 * 512];   // [h2, x1] padded K=256
  __shared__ __align__(32) _Float16 o3f[2 * 7 * 512];   // out3 padded K=224

  const int tid  = threadIdx.x;
  const int lane = tid & 31;
  const int wave = tid >> 5;                 // 0..15
  const int bm0  = blockIdx.x * BT;
  const int cf   = *cfp;

  for (int i = tid; i < BT * H; i += NTHREADS) { c1s[i] = 0.f; c2s[i] = 0.f; }
  for (int i = tid; i < 2 * 7 * 512; i += NTHREADS) {
    h1f[i] = (_Float16)0.f; h2f[i] = (_Float16)0.f; o3f[i] = (_Float16)0.f;
  }
  for (int i = tid; i < 2 * 8 * 512; i += NTHREADS) {
    x2f[i] = (_Float16)0.f; lpf[i] = (_Float16)0.f;
  }
  for (int i = tid; i < 2 * 2 * 512; i += NTHREADS) x1f[i] = (_Float16)0.f;
  for (int i = tid; i < BT * DA; i += NTHREADS) {
    int m = i / DA, r = i % DA;
    st0raw[i] = actions[(size_t)(bm0 + m) * DA + r];   // actions[0]
  }
  for (int i = tid; i < BT * DT; i += NTHREADS) o4raw[i] = 0.f;
  __syncthreads();

  for (int t = 0; t < TSTEPS - 1; ++t) {
    // ---- stage step inputs ----
    for (int i = tid; i < BT * DT; i += NTHREADS) {
      int m = i / DT, j = i % DT;
      x1raw[i] = (t < cf) ? tactiles[((size_t)t * Bk + bm0 + m) * DT + j] : o4raw[i];
    }
    for (int i = tid; i < BT * DA; i += NTHREADS) {
      int m = i / DA, r = i % DA;
      actraw[i] = actions[((size_t)(t + 1) * Bk + bm0 + m) * DA + r];
    }
    __syncthreads();
    for (int i = tid; i < BT * DT; i += NTHREADS) {
      int m = i / DT, j = i % DT;
      int mt = m >> 4, ml = m & 15;
      _Float16 xv = (_Float16)x1raw[i];
      x1f[mt * 2 * 512 + afrag_idx(ml, j)]       = xv;
      lpf[mt * 8 * 512 + afrag_idx(ml, 200 + j)] = xv;   // lp = [h2, x1]
      int r = j % 12;
      float tv = (r < 6) ? actraw[m * DA + r] : st0raw[m * DA + (r - 6)];
      x2f[mt * 8 * 512 + afrag_idx(ml, 200 + j)] = (_Float16)tv; // x2 = [h1, tiled]
    }
    __syncthreads();

    // ---- LSTM1: z = x1*Wih1^T + h1*Whh1^T + bb1 ----
    for (int nt = wave; nt < 50; nt += 16) {
      int n = nt * 16 + (lane & 15);
      float bv = bb1[n];
      v8f acc0, acc1;
      for (int v = 0; v < 8; ++v) { acc0[v] = bv; acc1[v] = bv; }
      wmma_k2(x1f, 2 * 512, Wih1f + (size_t)nt * 2 * 512, 2, acc0, acc1, lane);
      wmma_k2(h1f, 7 * 512, Whh1f + (size_t)nt * 7 * 512, 7, acc0, acc1, lane);
      int mh = 8 * (lane >> 4);
      for (int v = 0; v < 8; ++v) {
        zbuf[(mh + v) * 800 + n]      = acc0[v];
        zbuf[(16 + mh + v) * 800 + n] = acc1[v];
      }
    }
    __syncthreads();
    for (int i = tid; i < BT * H; i += NTHREADS) {
      int m = i / H, jj = i % H;
      const float* zr = zbuf + m * 800;
      float c = sigf(zr[200 + jj]) * c1s[i] + sigf(zr[jj]) * tanhf(zr[400 + jj]);
      float h = sigf(zr[600 + jj]) * tanhf(c);
      c1s[i] = c;
      _Float16 hh = (_Float16)h;
      int mt = m >> 4, fi = afrag_idx(m & 15, jj);
      h1f[mt * 7 * 512 + fi] = hh;   // recurrent input for next step
      x2f[mt * 8 * 512 + fi] = hh;   // feeds LSTM2 this step
    }
    __syncthreads();

    // ---- LSTM2: z = x2*Wih2^T + h2*Whh2^T + bb2 ----
    for (int nt = wave; nt < 50; nt += 16) {
      int n = nt * 16 + (lane & 15);
      float bv = bb2[n];
      v8f acc0, acc1;
      for (int v = 0; v < 8; ++v) { acc0[v] = bv; acc1[v] = bv; }
      wmma_k2(x2f, 8 * 512, Wih2f + (size_t)nt * 8 * 512, 8, acc0, acc1, lane);
      wmma_k2(h2f, 7 * 512, Whh2f + (size_t)nt * 7 * 512, 7, acc0, acc1, lane);
      int mh = 8 * (lane >> 4);
      for (int v = 0; v < 8; ++v) {
        zbuf[(mh + v) * 800 + n]      = acc0[v];
        zbuf[(16 + mh + v) * 800 + n] = acc1[v];
      }
    }
    __syncthreads();
    for (int i = tid; i < BT * H; i += NTHREADS) {
      int m = i / H, jj = i % H;
      const float* zr = zbuf + m * 800;
      float c = sigf(zr[200 + jj]) * c2s[i] + sigf(zr[jj]) * tanhf(zr[400 + jj]);
      float h = sigf(zr[600 + jj]) * tanhf(c);
      c2s[i] = c;
      _Float16 hh = (_Float16)h;
      int mt = m >> 4, fi = afrag_idx(m & 15, jj);
      h2f[mt * 7 * 512 + fi] = hh;
      lpf[mt * 8 * 512 + fi] = hh;
    }
    __syncthreads();

    // ---- FC1: out3 = tanh(lp*W1^T + b1) ----
    for (int nt = wave; nt < 13; nt += 16) {
      int n = nt * 16 + (lane & 15);
      float bv = b1p[n];
      v8f acc0, acc1;
      for (int v = 0; v < 8; ++v) { acc0[v] = bv; acc1[v] = bv; }
      wmma_k2(lpf, 8 * 512, W1f + (size_t)nt * 8 * 512, 8, acc0, acc1, lane);
      if (n < 200) {
        int mh = 8 * (lane >> 4);
        for (int v = 0; v < 8; ++v) {
          o3f[afrag_idx(mh + v, n)]           = (_Float16)tanhf(acc0[v]);
          o3f[7 * 512 + afrag_idx(mh + v, n)] = (_Float16)tanhf(acc1[v]);
        }
      }
    }
    __syncthreads();

    // ---- FC2: out4 = tanh(out3*W2^T + b2) ----
    for (int nt = wave; nt < 3; nt += 16) {
      int n = nt * 16 + (lane & 15);
      float bv = b2p[n];
      v8f acc0, acc1;
      for (int v = 0; v < 8; ++v) { acc0[v] = bv; acc1[v] = bv; }
      wmma_k2(o3f, 7 * 512, W2f + (size_t)nt * 7 * 512, 7, acc0, acc1, lane);
      int mh = 8 * (lane >> 4);
      for (int v = 0; v < 8; ++v) {
        o4raw[(mh + v) * DT + n]        = tanhf(acc0[v]);
        o4raw[(16 + mh + v) * DT + n]   = tanhf(acc1[v]);
      }
    }
    __syncthreads();

    // ---- emit output rows: last context frame + predictions ----
    if (t >= cf - 1) {
      int r = t - (cf - 1);
      for (int i = tid; i < BT * DT; i += NTHREADS) {
        int m = i / DT, j = i % DT;
        out[((size_t)r * Bk + bm0 + m) * DT + j] = o4raw[i];
      }
    }
    __syncthreads();
  }
}

extern "C" void kernel_launch(void* const* d_in, const int* in_sizes, int n_in,
                              void* d_out, int out_size, void* d_ws, size_t ws_size,
                              hipStream_t stream) {
  const float* tactiles = (const float*)d_in[0];
  const float* actions  = (const float*)d_in[1];
  const float* Wih1 = (const float*)d_in[2];
  const float* Whh1 = (const float*)d_in[3];
  const float* bih1 = (const float*)d_in[4];
  const float* bhh1 = (const float*)d_in[5];
  const float* Wih2 = (const float*)d_in[6];
  const float* Whh2 = (const float*)d_in[7];
  const float* bih2 = (const float*)d_in[8];
  const float* bhh2 = (const float*)d_in[9];
  const float* W1 = (const float*)d_in[10];
  const float* b1 = (const float*)d_in[11];
  const float* W2 = (const float*)d_in[12];
  const float* b2 = (const float*)d_in[13];
  const int*   cfp = (const int*)d_in[14];

  int Bk = in_sizes[0] / (TSTEPS * DT);   // 32768

  // workspace carve-up (byte offsets, 32B-aligned fragments)
  char* w = (char*)d_ws;
  _Float16* Wih1f = (_Float16*)(w + 0);        //  50*2*512 h = 102400 B
  _Float16* Whh1f = (_Float16*)(w + 102400);   //  50*7*512 h = 358400 B
  _Float16* Wih2f = (_Float16*)(w + 460800);   //  50*8*512 h = 409600 B
  _Float16* Whh2f = (_Float16*)(w + 870400);   //  50*7*512 h = 358400 B
  _Float16* W1f   = (_Float16*)(w + 1228800);  //  13*8*512 h = 106496 B
  _Float16* W2f   = (_Float16*)(w + 1335296);  //   3*7*512 h =  21504 B
  float* bb1 = (float*)(w + 1356800);
  float* bb2 = bb1 + 800;
  float* b1p = bb2 + 800;
  float* b2p = b1p + 208;

  prep_wfrag<<<(51200  + 255) / 256, 256, 0, stream>>>(Wih1, Wih1f, 800, 48,  2, 51200);
  prep_wfrag<<<(179200 + 255) / 256, 256, 0, stream>>>(Whh1, Whh1f, 800, 200, 7, 179200);
  prep_wfrag<<<(204800 + 255) / 256, 256, 0, stream>>>(Wih2, Wih2f, 800, 248, 8, 204800);
  prep_wfrag<<<(179200 + 255) / 256, 256, 0, stream>>>(Whh2, Whh2f, 800, 200, 7, 179200);
  prep_wfrag<<<(53248  + 255) / 256, 256, 0, stream>>>(W1,   W1f,   200, 248, 8, 53248);
  prep_wfrag<<<(10752  + 255) / 256, 256, 0, stream>>>(W2,   W2f,   48,  200, 7, 10752);
  prep_bias<<<4, 256, 0, stream>>>(bih1, bhh1, bih2, bhh2, b1, b2, bb1, bb2, b1p, b2p);

  actp_main<<<Bk / BT, NTHREADS, 0, stream>>>(tactiles, actions, cfp,
      Wih1f, Whh1f, Wih2f, Whh2f, W1f, W2f, bb1, bb2, b1p, b2p,
      (float*)d_out, Bk);
}
